// NewLLMResidual_45896020525781
// MI455X (gfx1250) — compile-verified, matrix operand order checked by reading, not run on hardware
//
#include <hip/hip_runtime.h>
#include <hip/hip_bf16.h>
#include <math.h>

// ---------------------------------------------------------------------------
// NewLLMResidual on MI455X (gfx1250, wave32)
//
// Roofline reasoning:
//  * Recurrence: 4096 sequential layer-GEMMs [8x2048]x[2048x2048]. Weights
//    (128MB f32) fit in the 192MB L2; converted once to bf16 (64MB) => 32GB
//    of L2-resident streaming. Per layer-step: 8MB from L2, spread over
//    32 WGPs (32 blocks x 8 waves, 2-way K-split per 16-col N-tile),
//    synchronized per layer with a device-scope atomic barrier (the serial
//    cost inherent to a scan). M=8 < 16: WMMA rows 8..15 alias rows 0..7
//    (results discarded) so the A-fragment path is branch-free. Two
//    independent WMMA accumulator chains hide the bf16-WMMA RAW hazard.
//  * Projection: [4096x1024]x[1024x32000] bf16 WMMA; 524MB f32 output write
//    is the HBM floor (~23us @ 23.3TB/s). ctxs is exported as bf16 so proj
//    A-fragments are raw b128 loads (no cvt in the hot loop).
// ---------------------------------------------------------------------------

typedef __attribute__((ext_vector_type(16))) __bf16 v16bf;
typedef __attribute__((ext_vector_type(8)))  float  v8f;

#define VOCAB 32000
#define EMBED 1024
#define CTX   1024
#define HID   2048
#define NLAY  8
#define BB    8
#define TT    512

#define REC_BLOCKS 32u   // 256 waves = 128 N-tiles x 2 K-halves

// workspace layout (all offsets 256B aligned)
#define OFF_STATE 0ull                         // 2(parity)*2(c|t)*8*1024 f32 = 131072 B
#define OFF_BAR   131072ull                    // 2 x u32 barrier state (+pad)
#define OFF_X     (OFF_BAR + 256ull)           // x f32 [8][512][1024]   = 16777216 B
#define OFF_CTXS  (OFF_X + 16777216ull)        // ctxs bf16 [8][512][1024] = 8388608 B
#define OFF_WF    (OFF_CTXS + 8388608ull)      // fused Wc|Wt bf16 frags = 67108864 B
#define OFF_WO    (OFF_WF + 67108864ull)       // Wo bf16 frags          = 65536000 B
// total ~150.6 MB of d_ws

union AV { v16bf v; uint4 u[2]; };

// ---------------------------------------------------------------------------
// device-scope barrier across the recurrence blocks (generation/sense based)
// ---------------------------------------------------------------------------
__device__ inline void grid_barrier(unsigned* bar) {
  __threadfence();
  __syncthreads();
  if (threadIdx.x == 0) {
    unsigned gen = __atomic_load_n(bar + 1, __ATOMIC_ACQUIRE);
    if (atomicAdd(bar, 1u) == REC_BLOCKS - 1u) {
      __atomic_store_n(bar, 0u, __ATOMIC_RELAXED);
      __threadfence();
      atomicAdd(bar + 1, 1u);                  // release generation
    } else {
      while (__atomic_load_n(bar + 1, __ATOMIC_ACQUIRE) == gen) {
        __builtin_amdgcn_s_sleep(1);
      }
    }
  }
  __syncthreads();
}

// ---------------------------------------------------------------------------
// init: zero the recurrence state + barrier (ws is not re-zeroed by harness)
// ---------------------------------------------------------------------------
__global__ void init_kernel(float* state, unsigned* bar) {
  for (int i = threadIdx.x; i < 2 * 2 * 8 * 1024; i += 256) state[i] = 0.f;
  if (threadIdx.x < 2) bar[threadIdx.x] = 0u;
}

// ---------------------------------------------------------------------------
// embedding gather + LayerNorm -> x f32 [B,T,EMBED]; one block per (b,t) row
// ---------------------------------------------------------------------------
__global__ void __launch_bounds__(256) embed_ln_kernel(
    const int* __restrict__ ids, const float* __restrict__ emb,
    const float* __restrict__ gamma, const float* __restrict__ beta,
    float* __restrict__ x) {
  __shared__ float red[256];
  const int row = blockIdx.x;
  const int tid = threadIdx.x;
  const int tok = ids[row];
  const float* e = emb + (size_t)tok * EMBED;
  float v[4]; float s = 0.f, ss = 0.f;
#pragma unroll
  for (int i = 0; i < 4; ++i) { v[i] = e[tid + 256 * i]; s += v[i]; ss += v[i] * v[i]; }
  red[tid] = s; __syncthreads();
  for (int off = 128; off > 0; off >>= 1) { if (tid < off) red[tid] += red[tid + off]; __syncthreads(); }
  float mu = red[0] * (1.f / EMBED); __syncthreads();
  red[tid] = ss; __syncthreads();
  for (int off = 128; off > 0; off >>= 1) { if (tid < off) red[tid] += red[tid + off]; __syncthreads(); }
  float var = red[0] * (1.f / EMBED) - mu * mu;
  float rs = rsqrtf(var + 1e-5f);
#pragma unroll
  for (int i = 0; i < 4; ++i) {
    int c = tid + 256 * i;
    x[(size_t)row * EMBED + c] = (v[i] - mu) * rs * gamma[c] + beta[c];
  }
}

// ---------------------------------------------------------------------------
// Convert Wc|Wt (f32, [L][2048][1024] each) into bf16 WMMA B-fragment order:
//   dst[((l*128+nt)*64+kc)*32 + lane][e]  (each lane: 16 bf16 = 32B contig)
//   element e <-> K = kc*32 + 16*(e>>3) + 8*(lane>>4) + (e&7), N = nt*16+(lane&15)
// ---------------------------------------------------------------------------
__global__ void __launch_bounds__(256) convert_wf_kernel(
    const float* __restrict__ Wc, const float* __restrict__ Wt, __bf16* __restrict__ wf) {
  int id = blockIdx.x * 256 + threadIdx.x;   // 2,097,152 fragment-lanes
  int lane = id & 31;
  int kc   = (id >> 5) & 63;
  int nt   = (id >> 11) & 127;
  int l    = id >> 18;
  int n    = nt * 16 + (lane & 15);
  int half = lane >> 4;
  AV av;
#pragma unroll
  for (int e = 0; e < 16; ++e) {
    int klocal = 16 * (e >> 3) + 8 * half + (e & 7);
    int k = kc * 32 + klocal;
    float v = (n < CTX) ? Wc[((size_t)l * HID + k) * CTX + n]
                        : Wt[((size_t)l * HID + k) * EMBED + (n - CTX)];
    av.v[e] = (__bf16)v;
  }
  ((v16bf*)wf)[id] = av.v;
}

// Same fragment swizzle for Wo [1024][32000]: dst[(nt*32+kc)*32+lane]
__global__ void __launch_bounds__(256) convert_wo_kernel(
    const float* __restrict__ Wo, __bf16* __restrict__ wo) {
  int id = blockIdx.x * 256 + threadIdx.x;   // 2,048,000
  int lane = id & 31;
  int kc   = (id >> 5) & 31;
  int nt   = id >> 10;                       // 0..1999
  int n    = nt * 16 + (lane & 15);
  int half = lane >> 4;
  AV av;
#pragma unroll
  for (int e = 0; e < 16; ++e) {
    int klocal = 16 * (e >> 3) + 8 * half + (e & 7);
    int k = kc * 32 + klocal;
    av.v[e] = (__bf16)Wo[(size_t)k * VOCAB + n];
  }
  ((v16bf*)wo)[id] = av.v;
}

// ---------------------------------------------------------------------------
// Persistent recurrence kernel: 32 blocks x 8 waves.
//   wave pair (2i, 2i+1) owns N-tile nt = blockIdx*4+i; khalf = wave&1 splits
//   the K=2048 reduction; partial sums combined through LDS.
// state layout: state[((parity*2 + which)*8 + batch)*1024 + col], which 0=c 1=t
// ---------------------------------------------------------------------------
__global__ void __launch_bounds__(256, 1) recurrence_kernel(
    const float* __restrict__ x, const __bf16* __restrict__ wf,
    const float* __restrict__ bc, const float* __restrict__ bt,
    float* __restrict__ state, __bf16* __restrict__ ctxs,
    unsigned* __restrict__ bar) {
  __shared__ __align__(16) __bf16 hA[8 * HID];   // 32 KB activation stage
  __shared__ float partial[4][32][9];            // K-split partials (pad: no bank conflict)
  const int tid   = threadIdx.x;
  const int lane  = tid & 31;
  const int half  = lane >> 4;
  const int l15   = lane & 15;                   // A row / C-D column
  const int wave  = tid >> 5;
  const int khalf = wave & 1;                    // which K half
  const int tilei = wave >> 1;                   // 0..3 tile within block
  const int nt    = blockIdx.x * 4 + tilei;      // 0..127 N-tile
  const int gtid  = blockIdx.x * 256 + tid;      // 0..8191
  // WMMA rows 8..15 alias rows 0..7 (their outputs are never written back)
  const int arow  = l15 & 7;

  for (int step = 0; step < TT; ++step) {
    // t := LN(embed)[:, step, :] into parity-0 t region
    if (gtid < BB * EMBED) {
      int b = gtid >> 10, col = gtid & 1023;
      state[((0 * 2 + 1) * 8 + b) * 1024 + col] = x[((size_t)b * TT + step) * 1024 + col];
    }
    grid_barrier(bar);

    for (int l = 0; l < NLAY; ++l) {
      const int rp = l & 1, wp = rp ^ 1;
      // stage h = [c | t] rows 0..7 as bf16 into LDS
      for (int i = tid; i < 8 * HID; i += 256) {
        int b = i >> 11, k = i & 2047;
        int which = k >> 10, col = k & 1023;
        hA[i] = (__bf16)state[((rp * 2 + which) * 8 + b) * 1024 + col];
      }
      __syncthreads();

      const v16bf* wbase = (const v16bf*)wf + (size_t)(l * 128 + nt) * 64 * 32;
      const __bf16* arowp = &hA[arow * HID + 8 * half];
      v8f acc0 = {}, acc1 = {};
      const int kc0 = khalf * 32, kc1 = kc0 + 32;
      for (int kc = kc0; kc < kc1; kc += 2) {
        // even-kc chain
        AV a0;
        const __bf16* ap0 = arowp + kc * 32;
        a0.u[0] = *(const uint4*)(ap0);          // K runs are 16B-contiguous
        a0.u[1] = *(const uint4*)(ap0 + 16);
        v16bf b0 = wbase[kc * 32 + lane];        // 32B contig per lane (L2 stream)
        __builtin_prefetch((const void*)(wbase + (size_t)(kc + 2) * 32 + lane), 0, 3);
        acc0 = __builtin_amdgcn_wmma_f32_16x16x32_bf16(
            false, a0.v, false, b0, (short)0, acc0, false, false);
        // odd-kc chain (independent accumulator: hides WMMA RAW hazard)
        AV a1;
        const __bf16* ap1 = ap0 + 32;
        a1.u[0] = *(const uint4*)(ap1);
        a1.u[1] = *(const uint4*)(ap1 + 16);
        v16bf b1 = wbase[(kc + 1) * 32 + lane];
        __builtin_prefetch((const void*)(wbase + (size_t)(kc + 3) * 32 + lane), 0, 3);
        acc1 = __builtin_amdgcn_wmma_f32_16x16x32_bf16(
            false, a1.v, false, b1, (short)0, acc1, false, false);
      }
#pragma unroll
      for (int r = 0; r < 8; ++r) acc0[r] += acc1[r];

      // combine K halves through LDS, then residual+tanh writeback (rows 0..7)
      if (khalf == 1) {
#pragma unroll
        for (int r = 0; r < 8; ++r) partial[tilei][lane][r] = acc0[r];
      }
      __syncthreads();
      if (khalf == 0 && half == 0) {
        int gcol = nt * 16 + l15;
        int which = gcol >> 10, cc = gcol & 1023;
        const float* bias = (which == 0) ? (bc + l * CTX) : (bt + l * EMBED);
#pragma unroll
        for (int r = 0; r < 8; ++r) {
          float tot  = acc0[r] + partial[tilei][lane][r];
          float oldv = state[((rp * 2 + which) * 8 + r) * 1024 + cc];
          state[((wp * 2 + which) * 8 + r) * 1024 + cc] = oldv + tanhf(tot + bias[cc]);
        }
      }
      grid_barrier(bar);   // publishes layer output; also protects hA/partial WAR
    }
    // 8 parity flips -> final c is back in parity 0; export (as bf16) to ctxs
    if (gtid < BB * CTX) {
      int b = gtid >> 10, col = gtid & 1023;
      ctxs[((size_t)b * TT + step) * 1024 + col] =
          (__bf16)state[((0 * 2 + 0) * 8 + b) * 1024 + col];
    }
    // next iteration's t-init writes a disjoint region; its grid_barrier orders
  }
}

// ---------------------------------------------------------------------------
// Output projection: [4096,1024] x [1024,32000] + bo, bf16 WMMA, f32 accum.
// One wave = 16 rows x 64 cols (4 N-tiles share the A fragments; the 4
// independent accumulators also cover the WMMA->WMMA hazard distance).
// ---------------------------------------------------------------------------
__global__ void __launch_bounds__(256) proj_kernel(
    const __bf16* __restrict__ ctxs, const __bf16* __restrict__ wo,
    const float* __restrict__ bo, float* __restrict__ out) {
  const int lane = threadIdx.x & 31;
  const int half = lane >> 4;
  const int l15  = lane & 15;
  const int w    = blockIdx.x * 8 + (threadIdx.x >> 5);  // 0..127999
  const int mt   = w / 500;                              // 0..255 (M tiles)
  const int nq   = w % 500;                              // 0..499 (N quads)
  const int arow = mt * 16 + l15;                        // A global row

  const v16bf* wb0 = (const v16bf*)wo + (size_t)(nq * 4 + 0) * 32 * 32;
  const v16bf* wb1 = (const v16bf*)wo + (size_t)(nq * 4 + 1) * 32 * 32;
  const v16bf* wb2 = (const v16bf*)wo + (size_t)(nq * 4 + 2) * 32 * 32;
  const v16bf* wb3 = (const v16bf*)wo + (size_t)(nq * 4 + 3) * 32 * 32;

  v8f a0 = {}, a1 = {}, a2 = {}, a3 = {};
  for (int kc = 0; kc < 32; ++kc) {
    const __bf16* ap = ctxs + (size_t)arow * 1024 + kc * 32 + 8 * half;
    AV av;
    av.u[0] = *(const uint4*)(ap);
    av.u[1] = *(const uint4*)(ap + 16);
    a0 = __builtin_amdgcn_wmma_f32_16x16x32_bf16(false, av.v, false, wb0[kc * 32 + lane], (short)0, a0, false, false);
    a1 = __builtin_amdgcn_wmma_f32_16x16x32_bf16(false, av.v, false, wb1[kc * 32 + lane], (short)0, a1, false, false);
    a2 = __builtin_amdgcn_wmma_f32_16x16x32_bf16(false, av.v, false, wb2[kc * 32 + lane], (short)0, a2, false, false);
    a3 = __builtin_amdgcn_wmma_f32_16x16x32_bf16(false, av.v, false, wb3[kc * 32 + lane], (short)0, a3, false, false);
  }
  const int col0 = (nq * 4) * 16 + l15;     // C/D: column = lane&15
#pragma unroll
  for (int r = 0; r < 8; ++r) {
    size_t row = (size_t)(mt * 16 + r + 8 * half);       // C/D row = r + 8*half
    float* o = out + row * VOCAB;
    o[col0 +  0] = a0[r] + bo[col0 +  0];
    o[col0 + 16] = a1[r] + bo[col0 + 16];
    o[col0 + 32] = a2[r] + bo[col0 + 32];
    o[col0 + 48] = a3[r] + bo[col0 + 48];
  }
}

// ---------------------------------------------------------------------------
extern "C" void kernel_launch(void* const* d_in, const int* in_sizes, int n_in,
                              void* d_out, int out_size, void* d_ws, size_t ws_size,
                              hipStream_t stream) {
  const int*   ids   = (const int*)d_in[0];
  const float* emb   = (const float*)d_in[1];
  const float* gamma = (const float*)d_in[2];
  const float* beta  = (const float*)d_in[3];
  const float* Wc    = (const float*)d_in[4];
  const float* bc    = (const float*)d_in[5];
  const float* Wt    = (const float*)d_in[6];
  const float* bt    = (const float*)d_in[7];
  const float* Wo    = (const float*)d_in[8];
  const float* bo    = (const float*)d_in[9];
  float* out = (float*)d_out;

  char* ws = (char*)d_ws;
  float*    state = (float*)(ws + OFF_STATE);
  unsigned* bar   = (unsigned*)(ws + OFF_BAR);
  float*    x     = (float*)(ws + OFF_X);
  __bf16*   ctxs  = (__bf16*)(ws + OFF_CTXS);
  __bf16*   wf    = (__bf16*)(ws + OFF_WF);
  __bf16*   wo    = (__bf16*)(ws + OFF_WO);

  init_kernel<<<1, 256, 0, stream>>>(state, bar);
  embed_ln_kernel<<<BB * TT, 256, 0, stream>>>(ids, emb, gamma, beta, x);
  convert_wf_kernel<<<8192, 256, 0, stream>>>(Wc, Wt, wf);   // 2,097,152 threads
  convert_wo_kernel<<<8000, 256, 0, stream>>>(Wo, wo);       // 2,048,000 threads
  recurrence_kernel<<<REC_BLOCKS, 256, 0, stream>>>(x, wf, bc, bt, state, ctxs, bar);
  proj_kernel<<<16000, 256, 0, stream>>>(ctxs, wo, bo, out); // 128000 waves
}
